// SwinTransformerBlock_2001454760426
// MI455X (gfx1250) — matrix operand order repaired and checked by reading, not compiled, not run on hardware
//
#include <hip/hip_runtime.h>
#include <hip/hip_bf16.h>
#include <math.h>

typedef __attribute__((ext_vector_type(16))) _Float16 v16h;
typedef __attribute__((ext_vector_type(8)))  float    v8f;

#define BSZ    32
#define HDIM   56
#define CDIM   192
#define NHEADS 6
#define HD     32
#define WS     7
#define SHIFTS 3
#define NTOK   49          // tokens per window
#define TOTW   2048        // 32 * 64 windows
#define TOTTOK 100352      // TOTW * NTOK
#define MLPH   768
#define CPBH   256

#define AS3 __attribute__((address_space(3)))

// ---------------------------------------------------------------------------
// WMMA fragment loaders (wave32, 16x16x32 f16 -> f32)
// A 16x32: lane holds row M=lane&15, 16 contiguous K starting (lane>=16)*16.
// B 32x16: lane holds col N=lane&15, 16 contiguous K starting (lane>=16)*16.
// Tiles stored row-major [16][ld]: A rows = M, B tiles stored as B^T rows = N.
// ---------------------------------------------------------------------------
__device__ inline v16h frag_a_lds(const _Float16* base, int ld, int koff) {
  int lane = threadIdx.x & 31;
  const _Float16* p = base + (lane & 15) * ld + koff + ((lane >> 4) << 4);
  v16h f;
#pragma unroll
  for (int i = 0; i < 16; ++i) f[i] = p[i];
  return f;
}

// works for both LDS tiles and pre-converted f16 weights in global memory:
// 16 contiguous halves per lane (one 32B load), no per-use conversion.
__device__ inline v16h frag_b_f16(const _Float16* base, int ld, int koff) {
  int lane = threadIdx.x & 31;
  const _Float16* p = base + (size_t)(lane & 15) * ld + koff + ((lane >> 4) << 4);
  v16h f;
#pragma unroll
  for (int i = 0; i < 16; ++i) f[i] = p[i];
  return f;
}

#define WMMA(a, b, c) __builtin_amdgcn_wmma_f32_16x16x32_f16(false, (a), false, (b), (short)0, (c), false, false)

// gfx1250 async global->LDS DMA: 16B per enabled lane, tracked by ASYNCcnt.
__device__ inline void async_copy_b128(unsigned lds_off, const void* gaddr) {
  asm volatile("global_load_async_to_lds_b128 %0, %1, off"
               :: "v"(lds_off), "v"(gaddr) : "memory");
}
__device__ inline void async_wait0() {
  asm volatile("s_wait_asynccnt 0" ::: "memory");
}
__device__ inline unsigned lds_off_of(const void* p) {
  return (unsigned)(uintptr_t)(AS3 const char*)p;
}

// token (window-order) -> source/dest pixel through the cyclic shift
__device__ inline int token_pixel(int t) {
  int w = t / NTOK, n = t % NTOK;
  int b = w >> 6, wy = (w >> 3) & 7, wx = w & 7;
  int r = n / WS, c = n % WS;
  int gh = (wy * WS + r + SHIFTS) % HDIM;
  int gw = (wx * WS + c + SHIFTS) % HDIM;
  return b * (HDIM * HDIM) + gh * HDIM + gw;
}

// ---------------------------------------------------------------------------
// K0: one-shot f32 -> f16 weight conversion (removes per-block cvt + halves
// weight bytes pulled from L2 by the GEMM kernels)
// ---------------------------------------------------------------------------
__global__ void cvt_w_kernel(const float* __restrict__ qkv_w, const float* __restrict__ proj_w,
                             const float* __restrict__ fc1_w, const float* __restrict__ fc2_w,
                             _Float16* __restrict__ qkvh, _Float16* __restrict__ projh,
                             _Float16* __restrict__ fc1h, _Float16* __restrict__ fc2h) {
  int i = blockIdx.x * blockDim.x + threadIdx.x;
  if (i < 3 * CDIM * CDIM) qkvh[i] = (_Float16)qkv_w[i];
  if (i < CDIM * CDIM)     projh[i] = (_Float16)proj_w[i];
  if (i < MLPH * CDIM)     fc1h[i]  = (_Float16)fc1_w[i];
  if (i < CDIM * MLPH)     fc2h[i]  = (_Float16)fc2_w[i];
}

// ---------------------------------------------------------------------------
// K1: continuous relative position bias table -> rpb[6][49][49]
// ---------------------------------------------------------------------------
__device__ inline float cpb_coord(int v) {
  float s = (float)v * (8.0f / 6.0f);
  float sgn = (s > 0.f) ? 1.f : ((s < 0.f) ? -1.f : 0.f);
  return sgn * log2f(fabsf(s) + 1.0f) * (1.0f / 3.0f);
}

__global__ void rpb_kernel(const float* __restrict__ w1, const float* __restrict__ b1,
                           const float* __restrict__ w2, float* __restrict__ rpb) {
  __shared__ float tab_s[169][NHEADS];
  int tid = threadIdx.x;
  if (tid < 169) {
    float t0 = cpb_coord(tid / 13 - 6);
    float t1 = cpb_coord(tid % 13 - 6);
    float acc[NHEADS] = {0.f, 0.f, 0.f, 0.f, 0.f, 0.f};
    for (int hh = 0; hh < CPBH; ++hh) {
      float hv = w1[hh * 2] * t0 + w1[hh * 2 + 1] * t1 + b1[hh];
      hv = fmaxf(hv, 0.f);
#pragma unroll
      for (int h = 0; h < NHEADS; ++h) acc[h] += w2[h * CPBH + hh] * hv;
    }
#pragma unroll
    for (int h = 0; h < NHEADS; ++h) tab_s[tid][h] = acc[h];
  }
  __syncthreads();
  for (int e = tid; e < NHEADS * NTOK * NTOK; e += blockDim.x) {
    int h = e / (NTOK * NTOK);
    int ij = e % (NTOK * NTOK);
    int i = ij / NTOK, j = ij % NTOK;
    int di = i / WS - j / WS + 6;
    int dj = i % WS - j % WS + 6;
    float tv = tab_s[di * 13 + dj][h];
    rpb[e] = 16.0f / (1.0f + expf(-tv));
  }
}

// ---------------------------------------------------------------------------
// K2: QKV projection, BM=32 (2 M-tiles per wave share each B fragment).
// grid (9, 3136), 128 threads (4 waves). out layout: [window][head][n][d] f16
// ---------------------------------------------------------------------------
__global__ void qkv_kernel(const float* __restrict__ x, const _Float16* __restrict__ qkvh,
                           const float* __restrict__ q_bias, const float* __restrict__ v_bias,
                           _Float16* __restrict__ qd, _Float16* __restrict__ kd,
                           _Float16* __restrict__ vd) {
  __shared__ _Float16 Atile[32][CDIM];
  int tid = threadIdx.x;
  int mbase = blockIdx.y * 32;
  for (int e = tid; e < 32 * CDIM; e += 128) {
    int m = e / CDIM, c = e % CDIM;
    int p = token_pixel(mbase + m);
    Atile[m][c] = (_Float16)x[(size_t)p * CDIM + c];
  }
  __syncthreads();
  int wid = tid >> 5, lane = tid & 31;
  int n0 = blockIdx.x * 64 + wid * 16;
  const _Float16* wbase = qkvh + (size_t)n0 * CDIM;
  v8f acc0 = {}, acc1 = {};
#pragma unroll
  for (int ks = 0; ks < CDIM / 32; ++ks) {
    v16h b  = frag_b_f16(wbase, CDIM, ks * 32);
    v16h a0 = frag_a_lds(&Atile[0][0],  CDIM, ks * 32);
    v16h a1 = frag_a_lds(&Atile[16][0], CDIM, ks * 32);
    acc0 = WMMA(a0, b, acc0);
    acc1 = WMMA(a1, b, acc1);
  }
  int col = n0 + (lane & 15);
  float bias = (col < CDIM) ? q_bias[col] : ((col >= 2 * CDIM) ? v_bias[col - 2 * CDIM] : 0.f);
  int h = (col % CDIM) / HD, d = col & (HD - 1);
#pragma unroll
  for (int mt = 0; mt < 2; ++mt) {
#pragma unroll
    for (int rr = 0; rr < 8; ++rr) {
      int m = mt * 16 + rr + ((lane >> 4) << 3);
      int t = mbase + m;
      int w = t / NTOK, n = t % NTOK;
      float av = mt ? acc1[rr] : acc0[rr];
      _Float16 hv = (_Float16)(av + bias);
      size_t off = ((size_t)(w * NHEADS + h) * NTOK + n) * HD + d;
      if (col < CDIM) qd[off] = hv;
      else if (col < 2 * CDIM) kd[off] = hv;
      else vd[off] = hv;
    }
  }
}

// ---------------------------------------------------------------------------
// K3: cosine-attention row normalization; logit scale folded into q.
// ---------------------------------------------------------------------------
__global__ void normalize_qk(_Float16* __restrict__ qd, _Float16* __restrict__ kd,
                             const float* __restrict__ logit_scale) {
  int row = blockIdx.x * blockDim.x + threadIdx.x;
  const int total = TOTW * NHEADS * NTOK;
  if (row >= 2 * total) return;
  bool isq = row < total;
  int r = isq ? row : row - total;
  _Float16* ptr = (isq ? qd : kd) + (size_t)r * HD;
  float v[HD];
  float ss = 0.f;
#pragma unroll
  for (int i = 0; i < HD; ++i) { v[i] = (float)ptr[i]; ss += v[i] * v[i]; }
  float inv = 1.0f / fmaxf(sqrtf(ss), 1e-12f);
  if (isq) {
    int h = (r / NTOK) % NHEADS;
    inv *= expf(fminf(logit_scale[h], 4.60517019f));  // ln(100)
  }
#pragma unroll
  for (int i = 0; i < HD; ++i) ptr[i] = (_Float16)(v[i] * inv);
}

// ---------------------------------------------------------------------------
// K4: attention per (window, head). grid 12288, 128 threads (4 waves).
// Q/K row blocks staged with async global->LDS DMA (no VGPR round-trip);
// transposed V and zero padding staged normally.
// ---------------------------------------------------------------------------
__device__ inline int reg3(int y) { return (y < 49) ? 0 : ((y < 53) ? 1 : 2); }

__global__ void attention_kernel(const _Float16* __restrict__ qd, const _Float16* __restrict__ kd,
                                 const _Float16* __restrict__ vd, const float* __restrict__ rpb,
                                 _Float16* __restrict__ attn_out) {
  __shared__ _Float16 Qs[64][HD];
  __shared__ _Float16 Ks[64][HD];
  __shared__ _Float16 Vt[HD][64];
  __shared__ float    Ss[64][64];
  __shared__ _Float16 Ps[64][64];
  int tid = threadIdx.x;
  int w = blockIdx.x / NHEADS, h = blockIdx.x % NHEADS;
  size_t base = (size_t)(w * NHEADS + h) * NTOK * HD;
  {
    const _Float16* gq = qd + base;
    const _Float16* gk = kd + base;
    unsigned lq = lds_off_of(&Qs[0][0]);
    unsigned lk = lds_off_of(&Ks[0][0]);
    for (int e = tid; e < (NTOK * HD) / 8; e += 128) {   // 196 x 16B chunks
      async_copy_b128(lq + e * 16, gq + e * 8);
      async_copy_b128(lk + e * 16, gk + e * 8);
    }
    for (int e = tid; e < 64 * HD; e += 128) {
      int r = e >> 5, d = e & (HD - 1);
      if (r >= NTOK) { Qs[r][d] = (_Float16)0.f; Ks[r][d] = (_Float16)0.f; }
      Vt[d][r] = (r < NTOK) ? vd[base + r * HD + d] : (_Float16)0.f;
    }
    async_wait0();
  }
  __syncthreads();
  int wid = tid >> 5, lane = tid & 31;
  int wy = (w & 63) >> 3, wx = w & 7;
  // S = q_hat @ k_hat^T (K = 32, single WMMA per 16x16 tile) + rpb + shift mask
  v16h aq = frag_a_lds(&Qs[wid * 16][0], HD, 0);   // invariant across nt
#pragma unroll
  for (int nt = 0; nt < 4; ++nt) {
    v16h b = frag_b_f16(&Ks[nt * 16][0], HD, 0);
    v8f acc = {};
    acc = WMMA(aq, b, acc);
    int col = nt * 16 + (lane & 15);
#pragma unroll
    for (int rr = 0; rr < 8; ++rr) {
      int row = wid * 16 + rr + ((lane >> 4) << 3);
      float val = -1e30f;
      if (row < NTOK && col < NTOK) {
        int ri = reg3(wy * WS + row / WS) * 3 + reg3(wx * WS + row % WS);
        int rj = reg3(wy * WS + col / WS) * 3 + reg3(wx * WS + col % WS);
        float msk = (ri != rj) ? -100.f : 0.f;
        val = acc[rr] + rpb[(h * NTOK + row) * NTOK + col] + msk;
      }
      Ss[row][col] = val;
    }
  }
  __syncthreads();
  if (tid < 64) {
    float mx = -1e30f;
    for (int j = 0; j < 64; ++j) mx = fmaxf(mx, Ss[tid][j]);
    float sum = 0.f;
    for (int j = 0; j < 64; ++j) sum += expf(Ss[tid][j] - mx);
    float inv = 1.0f / sum;
    for (int j = 0; j < 64; ++j) Ps[tid][j] = (_Float16)(expf(Ss[tid][j] - mx) * inv);
  }
  __syncthreads();
  // O = P @ V  (49x64pad @ 64x32)
#pragma unroll
  for (int nt = 0; nt < 2; ++nt) {
    v8f acc = {};
#pragma unroll
    for (int ks = 0; ks < 2; ++ks) {
      v16h a = frag_a_lds(&Ps[wid * 16][0], 64, ks * 32);
      v16h b = frag_b_f16(&Vt[nt * 16][0], 64, ks * 32);
      acc = WMMA(a, b, acc);
    }
    int col = nt * 16 + (lane & 15);
#pragma unroll
    for (int rr = 0; rr < 8; ++rr) {
      int row = wid * 16 + rr + ((lane >> 4) << 3);
      if (row < NTOK)
        attn_out[(size_t)(w * NTOK + row) * CDIM + h * HD + col] = (_Float16)acc[rr];
    }
  }
}

// ---------------------------------------------------------------------------
// K5: out-proj + window-reverse scatter + LayerNorm + residual -> x1 (f32)
// BM=32, grid 3136, 128 threads. A tile staged via async global->LDS DMA.
// ---------------------------------------------------------------------------
__global__ void proj_ln_kernel(const _Float16* __restrict__ attn_out, const _Float16* __restrict__ projh,
                               const float* __restrict__ proj_b, const float* __restrict__ x,
                               const float* __restrict__ g, const float* __restrict__ bb,
                               float* __restrict__ x1) {
  __shared__ _Float16 Atile[32][CDIM];
  __shared__ float    Otile[32][CDIM];
  __shared__ int      pidx[32];
  __shared__ float    mu_s[32], rs_s[32];
  int tid = threadIdx.x;
  int mbase = blockIdx.x * 32;
  {
    const _Float16* gsrc = attn_out + (size_t)mbase * CDIM;
    unsigned la = lds_off_of(&Atile[0][0]);
    for (int e = tid; e < (32 * CDIM) / 8; e += 128) {   // 768 x 16B chunks
      async_copy_b128(la + e * 16, gsrc + e * 8);
    }
    if (tid < 32) pidx[tid] = token_pixel(mbase + tid);
    async_wait0();
  }
  __syncthreads();
  int wid = tid >> 5, lane = tid & 31;
#pragma unroll
  for (int j = 0; j < 3; ++j) {
    int n0 = wid * 16 + j * 64;
    const _Float16* wbase = projh + (size_t)n0 * CDIM;
    v8f acc0 = {}, acc1 = {};
#pragma unroll
    for (int ks = 0; ks < CDIM / 32; ++ks) {
      v16h b  = frag_b_f16(wbase, CDIM, ks * 32);
      v16h a0 = frag_a_lds(&Atile[0][0],  CDIM, ks * 32);
      v16h a1 = frag_a_lds(&Atile[16][0], CDIM, ks * 32);
      acc0 = WMMA(a0, b, acc0);
      acc1 = WMMA(a1, b, acc1);
    }
    int col = n0 + (lane & 15);
    float bias = proj_b[col];
#pragma unroll
    for (int rr = 0; rr < 8; ++rr) {
      int m = rr + ((lane >> 4) << 3);
      Otile[m][col]      = acc0[rr] + bias;
      Otile[m + 16][col] = acc1[rr] + bias;
    }
  }
  __syncthreads();
  if (tid < 32) {
    float s = 0.f;
    for (int c = 0; c < CDIM; ++c) s += Otile[tid][c];
    float mu = s * (1.0f / CDIM);
    float s2 = 0.f;
    for (int c = 0; c < CDIM; ++c) { float d = Otile[tid][c] - mu; s2 += d * d; }
    mu_s[tid] = mu;
    rs_s[tid] = rsqrtf(s2 * (1.0f / CDIM) + 1e-5f);
  }
  __syncthreads();
  for (int e = tid; e < 32 * CDIM; e += 128) {
    int m = e / CDIM, c = e % CDIM;
    size_t p = (size_t)pidx[m] * CDIM + c;
    x1[p] = x[p] + (Otile[m][c] - mu_s[m]) * rs_s[m] * g[c] + bb[c];
  }
}

// ---------------------------------------------------------------------------
// K6: fused MLP, BM=32: silu(x1@fc1^T+b1)@fc2^T+b2 -> LayerNorm -> + x1 -> out
// Hidden tile (32x768 f16) lives entirely in LDS; Otile overlays Htile after
// the fc2 accumulation completes (guarded by __syncthreads).
// ---------------------------------------------------------------------------
__global__ void mlp_kernel(const float* __restrict__ x1, const _Float16* __restrict__ fc1h,
                           const float* __restrict__ fc1_b, const _Float16* __restrict__ fc2h,
                           const float* __restrict__ fc2_b, const float* __restrict__ g2,
                           const float* __restrict__ b2, float* __restrict__ out) {
  __shared__ __align__(16) char smem[32 * CDIM * 2 + 32 * MLPH * 2];  // 12288 + 49152
  _Float16* Xtile = (_Float16*)smem;                     // [32][192]
  _Float16* Htile = (_Float16*)(smem + 32 * CDIM * 2);   // [32][768]
  float*    Otile = (float*)(smem + 32 * CDIM * 2);      // [32][192], aliases Htile
  __shared__ float mu_s[32], rs_s[32];
  int tid = threadIdx.x;
  int mbase = blockIdx.x * 32;
  for (int e = tid; e < 32 * CDIM; e += 128) {
    int m = e / CDIM, c = e % CDIM;
    Xtile[m * CDIM + c] = (_Float16)x1[(size_t)(mbase + m) * CDIM + c];
  }
  __syncthreads();
  int wid = tid >> 5, lane = tid & 31;
  // fc1 + SiLU : 48 column tiles of 16, 12 per wave, 2 M-tiles each
#pragma unroll
  for (int j = 0; j < 12; ++j) {
    int n0 = (wid + 4 * j) * 16;
    const _Float16* wbase = fc1h + (size_t)n0 * CDIM;
    v8f acc0 = {}, acc1 = {};
#pragma unroll
    for (int ks = 0; ks < CDIM / 32; ++ks) {
      v16h b  = frag_b_f16(wbase, CDIM, ks * 32);
      v16h a0 = frag_a_lds(Xtile,             CDIM, ks * 32);
      v16h a1 = frag_a_lds(Xtile + 16 * CDIM, CDIM, ks * 32);
      acc0 = WMMA(a0, b, acc0);
      acc1 = WMMA(a1, b, acc1);
    }
    int col = n0 + (lane & 15);
    float bias = fc1_b[col];
#pragma unroll
    for (int rr = 0; rr < 8; ++rr) {
      int m = rr + ((lane >> 4) << 3);
      float v0 = acc0[rr] + bias;
      float v1 = acc1[rr] + bias;
      Htile[m * MLPH + col]        = (_Float16)(v0 / (1.0f + expf(-v0)));  // silu
      Htile[(m + 16) * MLPH + col] = (_Float16)(v1 / (1.0f + expf(-v1)));
    }
  }
  __syncthreads();
  // fc2 : accumulate all tiles in registers first (Htile still live)
  v8f acc[6];
  {
    v8f z = {};
#pragma unroll
    for (int i = 0; i < 6; ++i) acc[i] = z;
  }
#pragma unroll
  for (int j = 0; j < 3; ++j) {
    int n0 = (wid + 4 * j) * 16;
    const _Float16* wbase = fc2h + (size_t)n0 * MLPH;
#pragma unroll
    for (int ks = 0; ks < MLPH / 32; ++ks) {
      v16h b  = frag_b_f16(wbase, MLPH, ks * 32);
      v16h a0 = frag_a_lds(Htile,             MLPH, ks * 32);
      v16h a1 = frag_a_lds(Htile + 16 * MLPH, MLPH, ks * 32);
      acc[2 * j]     = WMMA(a0, b, acc[2 * j]);
      acc[2 * j + 1] = WMMA(a1, b, acc[2 * j + 1]);
    }
  }
  __syncthreads();  // all waves done reading Htile -> safe to overlay Otile
#pragma unroll
  for (int j = 0; j < 3; ++j) {
    int n0 = (wid + 4 * j) * 16;
    int col = n0 + (lane & 15);
    float bias = fc2_b[col];
#pragma unroll
    for (int rr = 0; rr < 8; ++rr) {
      int m = rr + ((lane >> 4) << 3);
      Otile[m * CDIM + col]        = acc[2 * j][rr] + bias;
      Otile[(m + 16) * CDIM + col] = acc[2 * j + 1][rr] + bias;
    }
  }
  __syncthreads();
  if (tid < 32) {
    float s = 0.f;
    for (int c = 0; c < CDIM; ++c) s += Otile[tid * CDIM + c];
    float mu = s * (1.0f / CDIM);
    float s2 = 0.f;
    for (int c = 0; c < CDIM; ++c) { float d = Otile[tid * CDIM + c] - mu; s2 += d * d; }
    mu_s[tid] = mu;
    rs_s[tid] = rsqrtf(s2 * (1.0f / CDIM) + 1e-5f);
  }
  __syncthreads();
  for (int e = tid; e < 32 * CDIM; e += 128) {
    int m = e / CDIM, c = e % CDIM;
    size_t p = (size_t)(mbase + m) * CDIM + c;
    out[p] = x1[p] + (Otile[m * CDIM + c] - mu_s[m]) * rs_s[m] * g2[c] + b2[c];
  }
}

// ---------------------------------------------------------------------------
extern "C" void kernel_launch(void* const* d_in, const int* in_sizes, int n_in,
                              void* d_out, int out_size, void* d_ws, size_t ws_size,
                              hipStream_t stream) {
  (void)in_sizes; (void)n_in; (void)out_size; (void)ws_size;
  const float* x           = (const float*)d_in[0];
  const float* qkv_w       = (const float*)d_in[1];
  const float* q_bias      = (const float*)d_in[2];
  const float* v_bias      = (const float*)d_in[3];
  const float* logit_scale = (const float*)d_in[4];
  const float* cpb_w1      = (const float*)d_in[5];
  const float* cpb_b1      = (const float*)d_in[6];
  const float* cpb_w2      = (const float*)d_in[7];
  const float* proj_w      = (const float*)d_in[8];
  const float* proj_b      = (const float*)d_in[9];
  const float* n1g         = (const float*)d_in[10];
  const float* n1b         = (const float*)d_in[11];
  const float* n2g         = (const float*)d_in[12];
  const float* n2b         = (const float*)d_in[13];
  const float* fc1_w       = (const float*)d_in[14];
  const float* fc1_b       = (const float*)d_in[15];
  const float* fc2_w       = (const float*)d_in[16];
  const float* fc2_b       = (const float*)d_in[17];
  float* out = (float*)d_out;

  // workspace carve-up (~233 MB total)
  const size_t QKV_E = (size_t)TOTW * NHEADS * NTOK * HD;   // 19,267,584
  char* ws = (char*)d_ws;
  _Float16* qd = (_Float16*)ws;
  _Float16* kd = qd + QKV_E;
  _Float16* vd = kd + QKV_E;
  float* rpb = (float*)(vd + QKV_E);
  _Float16* attn = (_Float16*)((char*)rpb + (size_t)NHEADS * NTOK * NTOK * sizeof(float));
  float* x1 = (float*)(attn + (size_t)TOTTOK * CDIM);
  char* wend = (char*)(x1 + (size_t)TOTTOK * CDIM);
  wend += (16 - ((uintptr_t)wend & 15)) & 15;               // 16B align
  _Float16* qkvh = (_Float16*)wend;                         // 110592
  _Float16* projh = qkvh + 3 * CDIM * CDIM;                 // 36864
  _Float16* fc1h  = projh + CDIM * CDIM;                    // 147456
  _Float16* fc2h  = fc1h + MLPH * CDIM;                     // 147456

  cvt_w_kernel<<<(MLPH * CDIM + 255) / 256, 256, 0, stream>>>(
      qkv_w, proj_w, fc1_w, fc2_w, qkvh, projh, fc1h, fc2h);
  rpb_kernel<<<1, 256, 0, stream>>>(cpb_w1, cpb_b1, cpb_w2, rpb);
  qkv_kernel<<<dim3(9, TOTTOK / 32), 128, 0, stream>>>(x, qkvh, q_bias, v_bias, qd, kd, vd);
  {
    int rows = 2 * TOTW * NHEADS * NTOK;
    normalize_qk<<<(rows + 255) / 256, 256, 0, stream>>>(qd, kd, logit_scale);
  }
  attention_kernel<<<TOTW * NHEADS, 128, 0, stream>>>(qd, kd, vd, rpb, attn);
  proj_ln_kernel<<<TOTTOK / 32, 128, 0, stream>>>(attn, projh, proj_b, x, n1g, n1b, x1);
  mlp_kernel<<<TOTTOK / 32, 128, 0, stream>>>(x1, fc1h, fc1_b, fc2h, fc2_b, n2g, n2b, out);
}
